// HomogeneousNN_15968688406835
// MI455X (gfx1250) — matrix operand order repaired and checked by reading, not compiled, not run on hardware
//
#include <hip/hip_runtime.h>
#include <hip/hip_bf16.h>

// ---------------------------------------------------------------------------
// HomogeneousNN forward for MI455X (gfx1250, wave32, WMMA).
//   N=524288 samples, D=32, H=128, O=32, P = I (so ||y||_P is the L2 norm).
// Pipeline per wave (32 samples):
//   1) per-lane bisection solve  sum_d x_d^2 * exp(-2*lam_d*s) = 1   (fp32 VALU)
//   2) x_sphere = exp(-lam*S)*x  -> f16 LDS tile (A-matrix staging)
//   3) tanh MLP via v_wmma_f32_16x16x32_f16 (3 GEMMs, f32 accum, bias in C),
//      activations via hardware v_tanh_f32 where available
//   4) out = mlp * exp(S)
// ---------------------------------------------------------------------------

typedef __attribute__((ext_vector_type(16))) _Float16 v16h;
typedef __attribute__((ext_vector_type(8)))  _Float16 half8;
typedef __attribute__((ext_vector_type(2)))  _Float16 half2v;
typedef __attribute__((ext_vector_type(8)))  float    v8f;

#define D_IN  32
#define H_DIM 128
#define O_DIM 32
#define WAVES_PER_BLOCK 4
#define BLOCK_THREADS   (WAVES_PER_BLOCK * 32)
#define SAMPLES_PER_WAVE 32
#define SAMPLES_PER_BLOCK (WAVES_PER_BLOCK * SAMPLES_PER_WAVE)  // 128

// ---- dynamic LDS layout (bytes). All fragment reads are 16B aligned. -------
#define LAM_OFF   0                       // 32 f32
#define B0_OFF    128                     // 128 f32
#define B1_OFF    (B0_OFF + 512)          // 128 f32
#define W0T_OFF   (B1_OFF + 512)          // 1152 ; f16 [128][40]  (W0^T, pad)
#define W0T_STRIDE 40                     // 80B rows -> 16B aligned, bank-spread
#define W1T_OFF   (W0T_OFF + 128*W0T_STRIDE*2)   // +10240 ; f16 [128][136]
#define W1T_STRIDE 136                    // 272B rows
#define W2T_OFF   (W1T_OFF + 128*W1T_STRIDE*2)   // +34816 ; f16 [32][136]
#define W2T_STRIDE 136
#define WAVE_OFF  (W2T_OFF + 32*W2T_STRIDE*2)    // +8704 = 54912
#define XS_STRIDE 40                      // f16 [32][40] per wave (x_sphere)
#define XS_BYTES  (32*XS_STRIDE*2)        // 2560
#define H_STRIDE  136                     // f16 [16][136] per wave (h tile)
#define H_BYTES   (16*H_STRIDE*2)         // 4352
#define SCALE_BYTES 128                   // 32 f32 per wave
#define WAVE_BYTES (XS_BYTES + H_BYTES + SCALE_BYTES)            // 7040
#define SMEM_BYTES (WAVE_OFF + WAVES_PER_BLOCK * WAVE_BYTES)     // 83072

// Hardware tanh (CDNA5 lists TANH as a transcendental VALU op).
__device__ __forceinline__ float fast_tanh(float v) {
#if defined(__has_builtin)
#if __has_builtin(__builtin_amdgcn_tanhf)
  return __builtin_amdgcn_tanhf(v);
#elif __has_builtin(__builtin_amdgcn_tanh_f32)
  return __builtin_amdgcn_tanh_f32(v);
#else
  return tanhf(v);
#endif
#else
  return tanhf(v);
#endif
}

// Intra-wave LDS write->read fence: LDS is in-order per wave; the asm stops
// compiler reordering and drains DScnt for safety.
__device__ __forceinline__ void lds_fence() {
  asm volatile("s_wait_dscnt 0" ::: "memory");
}

// A-matrix fragment (16x32 f16): lane half j holds
//   K = kbase + j + (j>=8 ? 8 : 0) + (lane>=16 ? 8 : 0), row = lane&15
// -> two contiguous 8-half (16B) groups.
__device__ __forceinline__ v16h load_afrag(const _Float16* mat, int stride,
                                           int row, int kbase, int hi16) {
  const _Float16* p = mat + row * stride + kbase + hi16 * 8;
  half8 g0 = *(const half8*)p;
  half8 g1 = *(const half8*)(p + 16);
  return __builtin_shufflevector(g0, g1, 0,1,2,3,4,5,6,7,8,9,10,11,12,13,14,15);
}

// B-matrix fragment (32x16 f16) from transposed weights Wt[n][k]:
//   lane half j holds K = kbase + j + (j>=8 ? 8 : 0) + (lane>=16 ? 8 : 0), col n = lane&15
// -> two contiguous 8-half (16B) groups.
__device__ __forceinline__ v16h load_bfrag(const _Float16* wt, int stride,
                                           int n, int kbase, int hi16) {
  const _Float16* p = wt + n * stride + kbase + hi16 * 8;
  half8 g0 = *(const half8*)p;
  half8 g1 = *(const half8*)(p + 16);
  return __builtin_shufflevector(g0, g1, 0,1,2,3,4,5,6,7,8,9,10,11,12,13,14,15);
}

__device__ __forceinline__ v8f wmma_f16(v16h a, v16h b, v8f c) {
  return __builtin_amdgcn_wmma_f32_16x16x32_f16(false, a, false, b,
                                                (short)0, c, false, false);
}

__global__ void __launch_bounds__(BLOCK_THREADS)
homog_nn_kernel(const float* __restrict__ x, const float* __restrict__ lam,
                const float* __restrict__ W0, const float* __restrict__ b0,
                const float* __restrict__ W1, const float* __restrict__ b1,
                const float* __restrict__ W2, float* __restrict__ out, int n) {
  extern __shared__ char smem[];
  float*    lam_lds = (float*)(smem + LAM_OFF);
  float*    b0_lds  = (float*)(smem + B0_OFF);
  float*    b1_lds  = (float*)(smem + B1_OFF);
  _Float16* W0t     = (_Float16*)(smem + W0T_OFF);
  _Float16* W1t     = (_Float16*)(smem + W1T_OFF);
  _Float16* W2t     = (_Float16*)(smem + W2T_OFF);

  const int tid  = threadIdx.x;
  const int lane = tid & 31;
  const int wave = tid >> 5;
  const int row  = lane & 15;
  const int hi16 = (lane >> 4) & 1;

  char* wbase = smem + WAVE_OFF + wave * WAVE_BYTES;
  _Float16* xs        = (_Float16*)wbase;
  _Float16* hld       = (_Float16*)(wbase + XS_BYTES);
  float*    scale_lds = (float*)(wbase + XS_BYTES + H_BYTES);

  // ---- one-time cooperative weight staging (f32 -> f16, transposed) -------
  for (int i = tid; i < D_IN; i += BLOCK_THREADS) lam_lds[i] = lam[i];
  for (int i = tid; i < H_DIM; i += BLOCK_THREADS) {
    b0_lds[i] = b0[i];
    b1_lds[i] = b1[i];
  }
  for (int i = tid; i < D_IN * H_DIM; i += BLOCK_THREADS) {   // W0[32][128]
    int k = i & (D_IN - 1), nn = i >> 5;
    W0t[nn * W0T_STRIDE + k] = (_Float16)W0[k * H_DIM + nn];
  }
  for (int i = tid; i < H_DIM * H_DIM; i += BLOCK_THREADS) {  // W1[128][128]
    int k = i & (H_DIM - 1), nn = i >> 7;
    W1t[nn * W1T_STRIDE + k] = (_Float16)W1[k * H_DIM + nn];
  }
  for (int i = tid; i < H_DIM * O_DIM; i += BLOCK_THREADS) {  // W2[128][32]
    int k = i & (H_DIM - 1), nn = i >> 7;
    W2t[nn * W2T_STRIDE + k] = (_Float16)W2[k * O_DIM + nn];
  }
  __syncthreads();

  // lam in registers (invariant across tiles)
  float lam_r[D_IN];
  float alpha = 3.4e38f, beta = -3.4e38f;
#pragma unroll
  for (int d = 0; d < D_IN; ++d) {
    lam_r[d] = lam_lds[d];
    alpha = fminf(alpha, lam_r[d]);
    beta  = fmaxf(beta, lam_r[d]);
  }

  const int tiles = n / SAMPLES_PER_BLOCK;   // N is a multiple of 128
  for (int tile = blockIdx.x; tile < tiles; tile += gridDim.x) {
    const long base   = (long)tile * SAMPLES_PER_BLOCK + wave * SAMPLES_PER_WAVE;
    const long sample = base + lane;
    const float* xp   = x + sample * D_IN;

    // ---- 1) bisection: sum_d q_d * exp(-2*lam_d*s) = 1 --------------------
    float q[D_IN];
    float r2 = 0.f;
#pragma unroll
    for (int i = 0; i < 8; ++i) {
      float4 v = ((const float4*)xp)[i];
      q[4*i+0] = v.x * v.x; q[4*i+1] = v.y * v.y;
      q[4*i+2] = v.z * v.z; q[4*i+3] = v.w * v.w;
      r2 += q[4*i+0] + q[4*i+1] + q[4*i+2] + q[4*i+3];
    }
    const float logr = 0.5f * __logf(fmaxf(r2, 1e-30f));
    float lo = fminf(logr / alpha, logr / beta);
    float hi = fmaxf(logr / alpha, logr / beta);
    for (int it = 0; it < 1000; ++it) {
      if (!(hi - lo > 1e-4f)) break;
      const float mid = 0.5f * (lo + hi);
      float t = 0.f;
#pragma unroll
      for (int d = 0; d < D_IN; ++d)
        t += q[d] * __expf(-2.f * lam_r[d] * mid);
      if (t > 1.f) lo = mid; else hi = mid;
    }
    const float S = 0.5f * (lo + hi);
    scale_lds[lane] = __expf(S);        // NU = 1

    // ---- 2) x_sphere -> f16 LDS tile (row = lane) -------------------------
    _Float16* xr = xs + lane * XS_STRIDE;
#pragma unroll
    for (int i = 0; i < 8; ++i) {
      float4 v = ((const float4*)xp)[i];
      float e0 = __expf(-lam_r[4*i+0] * S), e1 = __expf(-lam_r[4*i+1] * S);
      float e2 = __expf(-lam_r[4*i+2] * S), e3 = __expf(-lam_r[4*i+3] * S);
      half2v p0 = { (_Float16)(v.x * e0), (_Float16)(v.y * e1) };
      half2v p1 = { (_Float16)(v.z * e2), (_Float16)(v.w * e3) };
      ((half2v*)xr)[2*i+0] = p0;
      ((half2v*)xr)[2*i+1] = p1;
    }
    lds_fence();

    // ---- 3) tanh MLP, two 16-sample M-tiles per wave ----------------------
    for (int mt = 0; mt < 2; ++mt) {
      // GEMM1: [16x32] @ [32x128] + b0, K fits one WMMA
      v16h a0 = load_afrag(xs, XS_STRIDE, mt * 16 + row, 0, hi16);
#pragma unroll
      for (int nt = 0; nt < 8; ++nt) {
        const float bv = b0_lds[nt * 16 + row];
        v8f c;
#pragma unroll
        for (int i = 0; i < 8; ++i) c[i] = bv;
        v16h b = load_bfrag(W0t, W0T_STRIDE, nt * 16 + row, 0, hi16);
        c = wmma_f16(a0, b, c);
#pragma unroll
        for (int i = 0; i < 8; ++i)   // D layout: M = i + hi16*8, N = row
          hld[(i + hi16 * 8) * H_STRIDE + nt * 16 + row] =
              (_Float16)fast_tanh(c[i]);
      }
      lds_fence();

      // GEMM2: [16x128] @ [128x128] + b1
      v8f acc2[8];
#pragma unroll
      for (int nt = 0; nt < 8; ++nt) {
        const float bv = b1_lds[nt * 16 + row];
#pragma unroll
        for (int i = 0; i < 8; ++i) acc2[nt][i] = bv;
      }
#pragma unroll
      for (int kc = 0; kc < 4; ++kc) {
        v16h a = load_afrag(hld, H_STRIDE, row, kc * 32, hi16);
#pragma unroll
        for (int nt = 0; nt < 8; ++nt) {
          v16h b = load_bfrag(W1t, W1T_STRIDE, nt * 16 + row, kc * 32, hi16);
          acc2[nt] = wmma_f16(a, b, acc2[nt]);
        }
      }
      lds_fence();   // all reads of h done before overwrite (WAR, in-order)
#pragma unroll
      for (int nt = 0; nt < 8; ++nt)
#pragma unroll
        for (int i = 0; i < 8; ++i)
          hld[(i + hi16 * 8) * H_STRIDE + nt * 16 + row] =
              (_Float16)fast_tanh(acc2[nt][i]);
      lds_fence();

      // GEMM3: [16x128] @ [128x32], then scale by exp(S)
      v16h a3[4];
#pragma unroll
      for (int kc = 0; kc < 4; ++kc)
        a3[kc] = load_afrag(hld, H_STRIDE, row, kc * 32, hi16);
      float sc[8];
#pragma unroll
      for (int i = 0; i < 8; ++i)
        sc[i] = scale_lds[mt * 16 + i + hi16 * 8];
#pragma unroll
      for (int nt = 0; nt < 2; ++nt) {
        v8f c;
#pragma unroll
        for (int i = 0; i < 8; ++i) c[i] = 0.f;
#pragma unroll
        for (int kc = 0; kc < 4; ++kc) {
          v16h b = load_bfrag(W2t, W2T_STRIDE, nt * 16 + row, kc * 32, hi16);
          c = wmma_f16(a3[kc], b, c);
        }
#pragma unroll
        for (int i = 0; i < 8; ++i) {
          const long m = base + mt * 16 + i + hi16 * 8;
          out[m * O_DIM + nt * 16 + row] = c[i] * sc[i];
        }
      }
    }
  }
}

extern "C" void kernel_launch(void* const* d_in, const int* in_sizes, int n_in,
                              void* d_out, int out_size, void* d_ws, size_t ws_size,
                              hipStream_t stream) {
  const float* x   = (const float*)d_in[0];
  const float* lam = (const float*)d_in[1];
  // d_in[2] = P (identity in this problem; norm reduces to L2) -- unused
  const float* W0  = (const float*)d_in[3];
  const float* b0  = (const float*)d_in[4];
  const float* W1  = (const float*)d_in[5];
  const float* b1  = (const float*)d_in[6];
  const float* W2  = (const float*)d_in[7];
  float* out = (float*)d_out;

  const int n = in_sizes[0] / D_IN;   // number of samples

  (void)hipFuncSetAttribute((const void*)homog_nn_kernel,
                            hipFuncAttributeMaxDynamicSharedMemorySize,
                            SMEM_BYTES);

  const int grid = 512;               // persistent blocks; each loops tiles
  homog_nn_kernel<<<grid, BLOCK_THREADS, SMEM_BYTES, stream>>>(
      x, lam, W0, b0, W1, b1, W2, out, n);
}
